// MicroTransformerLayer_41815801594658
// MI455X (gfx1250) — matrix operand order repaired and checked by reading, not compiled, not run on hardware
//
#include <hip/hip_runtime.h>
#include <hip/hip_bf16.h>
#include <math.h>

// ---------------------------------------------------------------------------
// MicroTransformerLayer for MI455X (gfx1250, wave32, WMMA).
// bf16 activations through v_wmma_f32_16x16x32_bf16 (fp32 accum), fp32
// residual stream, flash-style attention. Tile staging uses CDNA5 async
// global->LDS DMA (GLOBAL_LOAD_ASYNC_TO_LDS_B128, ASYNCcnt) with double
// buffering; fp32 x-tile is VGPR-staged (needs cvt) with global_prefetch_b8.
// ---------------------------------------------------------------------------

typedef __attribute__((ext_vector_type(16))) __bf16 v16bf;
typedef __attribute__((ext_vector_type(8)))  float  v8f;

__device__ __forceinline__ v8f wmma_bf16(v16bf a, v16bf b, v8f c) {
  // (neg_a, A, neg_b, B, c_mod, C, reuse_a, reuse_b)
  return __builtin_amdgcn_wmma_f32_16x16x32_bf16(false, a, false, b,
                                                 (short)0, c, false, false);
}

// Async 16B global->LDS copy per lane (tracked by ASYNCcnt).
__device__ __forceinline__ void async_copy16(const void* g, void* lds) {
  unsigned l = (unsigned)(size_t)lds;   // low 32 bits = LDS byte address
  asm volatile("global_load_async_to_lds_b128 %0, %1, off"
               :: "v"(l), "v"(g) : "memory");
}
template <int N>
__device__ __forceinline__ void wait_asynccnt() {
  asm volatile("s_wait_asynccnt %0" :: "n"(N) : "memory");
}

// ---------------------------------------------------------------------------
// Weight prep: W (N,K) fp32 -> Wt (K,N) bf16
// ---------------------------------------------------------------------------
__global__ __launch_bounds__(256) void wprep_kernel(const float* __restrict__ W,
                                                    __bf16* __restrict__ Wt,
                                                    int N, int K) {
  int idx = blockIdx.x * 256 + threadIdx.x;   // over K*N, write-coalesced
  int k = idx / N;
  int n = idx - k * N;
  Wt[idx] = (__bf16)W[(size_t)n * K + k];
}

// ---------------------------------------------------------------------------
// RMSNorm over rows of 256: one wave per row, 8 rows per block. bf16 out.
// ---------------------------------------------------------------------------
__global__ __launch_bounds__(256) void rms_kernel(const float* __restrict__ x,
                                                  const float* __restrict__ g,
                                                  __bf16* __restrict__ out) {
  int wave = threadIdx.x >> 5, lane = threadIdx.x & 31;
  size_t row = (size_t)blockIdx.x * 8 + wave;
  const float* xr = x + row * 256;
  float v[8];
  float ss = 0.f;
#pragma unroll
  for (int i = 0; i < 8; ++i) { v[i] = xr[lane + 32 * i]; ss += v[i] * v[i]; }
#pragma unroll
  for (int mk = 1; mk < 32; mk <<= 1) ss += __shfl_xor(ss, mk, 32);
  float r = rsqrtf(ss * (1.0f / 256.0f) + 1.1920929e-07f);
#pragma unroll
  for (int i = 0; i < 8; ++i) {
    int c = lane + 32 * i;
    out[row * 256 + c] = (__bf16)(v[i] * r * g[c]);
  }
}

// ---------------------------------------------------------------------------
// Generic GEMM: C(M,N) = A(M,K) @ Bt(K,N)  [Bt = W^T, bf16]
// Block tile: 128 x (64*NF), 8 waves; wave does 16 rows x 64*NF cols.
// AK: 0 = A fp32 (VGPR stage + cvt), 1 = A bf16 (async DMA stage).
// EPI: 0 f32 out | 1 bf16 out | 2 f32 out + residual | 3 bf16 out + residual
// B tiles: async global->LDS DMA, double buffered (ping/pong on ASYNCcnt).
// ---------------------------------------------------------------------------
template <int AK, int EPI, int NF>
__global__ __launch_bounds__(256) void gemm_kernel(
    const void* __restrict__ Ap, const __bf16* __restrict__ Bt,
    const float* __restrict__ Res, void* __restrict__ Cp,
    int M, int N, int K) {
  __shared__ __align__(16) __bf16 As[128][48];               // 96B rows
  __shared__ __align__(16) __bf16 Bs[2][32][64 * NF + 16];   // 16B-mult rows

  const int tid = threadIdx.x;
  const int wave = tid >> 5, lane = tid & 31;
  const int lane16 = lane & 15, hi = lane >> 4;
  const int mbase = blockIdx.y * 128;
  const int nbase = blockIdx.x * (64 * NF);

  const float*  Af = (const float*)Ap;
  const __bf16* Ab = (const __bf16*)Ap;

  v8f acc[4 * NF];
#pragma unroll
  for (int i = 0; i < 4 * NF; ++i) acc[i] = (v8f){};

  const int nk = K / 32;

  // Issue B tile 0 (NF async chunk-copies per thread).
#pragma unroll
  for (int i = 0; i < NF; ++i) {
    int chunk = tid + 256 * i;                 // 16B chunks: 32 rows x 8*NF
    int r = chunk / (8 * NF);
    int c16 = chunk - r * (8 * NF);
    async_copy16(Bt + (size_t)r * N + nbase + c16 * 8, &Bs[0][r][c16 * 8]);
  }

  for (int kt = 0; kt < nk; ++kt) {
    const int kb = kt * 32;
    const int buf = kt & 1;

    // Stage A tile (128 x 32).
    if (AK == 0) {
      int r = tid >> 1;
      int c0 = (tid & 1) * 16;
      size_t base = (size_t)(mbase + r) * K + kb + c0;
#pragma unroll
      for (int i = 0; i < 16; ++i) As[r][c0 + i] = (__bf16)Af[base + i];
      if (kb + 32 < K) __builtin_prefetch(&Af[base + 32], 0, 0);
    } else {
#pragma unroll
      for (int i = 0; i < 2; ++i) {
        int chunk = tid + 256 * i;             // 128 rows x 4 chunks
        int r = chunk >> 2;
        int c16 = chunk & 3;
        async_copy16(Ab + (size_t)(mbase + r) * K + kb + c16 * 8,
                     &As[r][c16 * 8]);
      }
    }
    // Issue next B tile into the other buffer.
    if (kt + 1 < nk) {
#pragma unroll
      for (int i = 0; i < NF; ++i) {
        int chunk = tid + 256 * i;
        int r = chunk / (8 * NF);
        int c16 = chunk - r * (8 * NF);
        async_copy16(Bt + (size_t)(kb + 32 + r) * N + nbase + c16 * 8,
                     &Bs[buf ^ 1][r][c16 * 8]);
      }
      wait_asynccnt<NF>();   // tile kt (A + B) resident; only next-B pending
    } else {
      wait_asynccnt<0>();
    }
    __syncthreads();

    // A fragment: lane holds row m = wave*16 + lane16;
    // lanes<16: K {0..7,16..23}, lanes>=16: K {8..15,24..31}.
    v16bf a;
    {
      int m = wave * 16 + lane16;
      int k0 = hi ? 8 : 0;
#pragma unroll
      for (int e = 0; e < 8; ++e) {
        a[e]     = As[m][k0 + e];
        a[e + 8] = As[m][16 + k0 + e];
      }
    }
#pragma unroll
    for (int nf = 0; nf < 4 * NF; ++nf) {
      v16bf b;   // lane = n column, 16 k values in table order
#pragma unroll
      for (int e = 0; e < 16; ++e) b[e] = Bs[buf][(e & 7) + hi * 8 + (e & 8) * 2][nf * 16 + lane16];
      acc[nf] = wmma_bf16(a, b, acc[nf]);
    }
    __syncthreads();
  }

  // Epilogue: C layout — element j, lane: row = j + 8*hi, col = lane16.
#pragma unroll
  for (int nf = 0; nf < 4 * NF; ++nf) {
#pragma unroll
    for (int j = 0; j < 8; ++j) {
      int r = mbase + wave * 16 + j + hi * 8;
      int c = nbase + nf * 16 + lane16;
      size_t idx = (size_t)r * N + c;
      float v = acc[nf][j];
      if (EPI == 2 || EPI == 3) v += Res[idx];
      if (EPI == 0 || EPI == 2) ((float*)Cp)[idx] = v;
      else                      ((__bf16*)Cp)[idx] = (__bf16)v;
    }
  }
}

// ---------------------------------------------------------------------------
// Fused gate/up GEMM with silu(g)*u epilogue. N = FF = 512.
// A async (single-buffer), Bg/Bu async double-buffered.
// ---------------------------------------------------------------------------
__global__ __launch_bounds__(256) void gemm_gateup_kernel(
    const __bf16* __restrict__ A, const __bf16* __restrict__ Bg,
    const __bf16* __restrict__ Bu, __bf16* __restrict__ C,
    int M, int N, int K) {
  __shared__ __align__(16) __bf16 As[128][48];
  __shared__ __align__(16) __bf16 Bgs[2][32][80];
  __shared__ __align__(16) __bf16 Bus[2][32][80];

  const int tid = threadIdx.x;
  const int wave = tid >> 5, lane = tid & 31;
  const int lane16 = lane & 15, hi = lane >> 4;
  const int mbase = blockIdx.y * 128;
  const int nbase = blockIdx.x * 64;

  v8f accg[4], accu[4];
#pragma unroll
  for (int i = 0; i < 4; ++i) { accg[i] = (v8f){}; accu[i] = (v8f){}; }

  const int nk = K / 32;
  {
    int r = tid >> 3, c16 = tid & 7;
    async_copy16(Bg + (size_t)r * N + nbase + c16 * 8, &Bgs[0][r][c16 * 8]);
    async_copy16(Bu + (size_t)r * N + nbase + c16 * 8, &Bus[0][r][c16 * 8]);
  }

  for (int kt = 0; kt < nk; ++kt) {
    const int kb = kt * 32;
    const int buf = kt & 1;
    // A tile via async DMA (2 chunks per thread).
#pragma unroll
    for (int i = 0; i < 2; ++i) {
      int chunk = tid + 256 * i;
      int r = chunk >> 2, c16 = chunk & 3;
      async_copy16(A + (size_t)(mbase + r) * K + kb + c16 * 8,
                   &As[r][c16 * 8]);
    }
    if (kt + 1 < nk) {
      int r = tid >> 3, c16 = tid & 7;
      async_copy16(Bg + (size_t)(kb + 32 + r) * N + nbase + c16 * 8,
                   &Bgs[buf ^ 1][r][c16 * 8]);
      async_copy16(Bu + (size_t)(kb + 32 + r) * N + nbase + c16 * 8,
                   &Bus[buf ^ 1][r][c16 * 8]);
      wait_asynccnt<2>();
    } else {
      wait_asynccnt<0>();
    }
    __syncthreads();

    v16bf a;
    {
      int m = wave * 16 + lane16;
      int k0 = hi ? 8 : 0;
#pragma unroll
      for (int e = 0; e < 8; ++e) {
        a[e]     = As[m][k0 + e];
        a[e + 8] = As[m][16 + k0 + e];
      }
    }
#pragma unroll
    for (int nf = 0; nf < 4; ++nf) {
      v16bf bg, bu;
#pragma unroll
      for (int e = 0; e < 16; ++e) {
        bg[e] = Bgs[buf][(e & 7) + hi * 8 + (e & 8) * 2][nf * 16 + lane16];
        bu[e] = Bus[buf][(e & 7) + hi * 8 + (e & 8) * 2][nf * 16 + lane16];
      }
      accg[nf] = wmma_bf16(a, bg, accg[nf]);
      accu[nf] = wmma_bf16(a, bu, accu[nf]);
    }
    __syncthreads();
  }

#pragma unroll
  for (int nf = 0; nf < 4; ++nf) {
#pragma unroll
    for (int j = 0; j < 8; ++j) {
      int r = mbase + wave * 16 + j + hi * 8;
      int c = nbase + nf * 16 + lane16;
      float gv = accg[nf][j];
      float uv = accu[nf][j];
      float sg = gv / (1.0f + __expf(-gv));   // silu
      C[(size_t)r * N + c] = (__bf16)(sg * uv);
    }
  }
}

// ---------------------------------------------------------------------------
// Flash attention: grid (T/128, B*H). Block = 128 q rows, 8 waves x 16 rows.
// qkv layout per token (768): [q(4x64) | k(4x64) | v(4x64)].
// K chunk transposed into LDS via VGPRs (DMA cannot transpose);
// V chunk staged via async DMA. Online softmax, P relaid via LDS scratch.
// ---------------------------------------------------------------------------
__global__ __launch_bounds__(256) void attn_kernel(const __bf16* __restrict__ qkv,
                                                   __bf16* __restrict__ out) {
  const int T = 2048;
  const int tid = threadIdx.x;
  const int wave = tid >> 5, lane = tid & 31;
  const int lane16 = lane & 15, hi = lane >> 4;
  const int b = blockIdx.y >> 2, hd = blockIdx.y & 3;
  const int qb = blockIdx.x * 128;

  __shared__ __align__(16) __bf16 KT[64][36];     // [d][k_local] (transposed)
  __shared__ __align__(16) __bf16 Vs[32][80];     // [k_local][d], 160B rows
  __shared__ __align__(16) __bf16 Ps[8][16][34];  // per-wave P scratch

  // Q fragments: A-layout, d 0..31 and 32..63.
  v16bf aQ0, aQ1;
  {
    int t = qb + wave * 16 + lane16;
    const __bf16* qp = qkv + ((size_t)(b * T + t)) * 768 + hd * 64;
    int k0 = hi ? 8 : 0;
#pragma unroll
    for (int e = 0; e < 8; ++e) {
      aQ0[e]     = qp[k0 + e];       aQ0[e + 8] = qp[16 + k0 + e];
      aQ1[e]     = qp[32 + k0 + e];  aQ1[e + 8] = qp[48 + k0 + e];
    }
  }

  v8f o[4];
#pragma unroll
  for (int i = 0; i < 4; ++i) o[i] = (v8f){};
  float mrow[8], lrow[8];
#pragma unroll
  for (int j = 0; j < 8; ++j) { mrow[j] = -1e30f; lrow[j] = 0.f; }

  const int nch = (qb + 128) / 32;
  for (int ch = 0; ch < nch; ++ch) {
    const int kc = ch * 32;
    // V chunk (32 x 64) via async DMA.
    {
      int r = tid >> 3, c16 = tid & 7;
      async_copy16(qkv + ((size_t)(b * T + kc + r)) * 768 + 512 + hd * 64 + c16 * 8,
                   &Vs[r][c16 * 8]);
    }
    // K chunk transposed into LDS via VGPRs.
    {
      int r = tid >> 3;             // k row 0..31
      int c0 = (tid & 7) * 8;       // d 0..56
      const __bf16* kp = qkv + ((size_t)(b * T + kc + r)) * 768 + 256 + hd * 64 + c0;
#pragma unroll
      for (int i = 0; i < 8; ++i) KT[c0 + i][r] = kp[i];
    }
    wait_asynccnt<0>();
    __syncthreads();

    // S = Q @ K^T : 2 n-frags (32 k-cols), 2 K-dim steps (d = 64).
    v8f s[2];
    s[0] = (v8f){}; s[1] = (v8f){};
#pragma unroll
    for (int nf = 0; nf < 2; ++nf) {
      v16bf b0, b1;
#pragma unroll
      for (int e = 0; e < 16; ++e) {
        b0[e] = KT[(e & 7) + hi * 8 + (e & 8) * 2][nf * 16 + lane16];        // d = table-order k
        b1[e] = KT[32 + (e & 7) + hi * 8 + (e & 8) * 2][nf * 16 + lane16];   // d = 32 + table-order k
      }
      s[nf] = wmma_bf16(aQ0, b0, s[nf]);
      s[nf] = wmma_bf16(aQ1, b1, s[nf]);
    }

    // Scale + causal mask + online softmax (per-row state in registers).
    float p[2][8];
#pragma unroll
    for (int j = 0; j < 8; ++j) {
      int q = qb + wave * 16 + j + hi * 8;
      float sm = -1e30f;
#pragma unroll
      for (int nf = 0; nf < 2; ++nf) {
        int k = kc + nf * 16 + lane16;
        float v = s[nf][j] * 0.125f;          // 1/sqrt(64)
        if (k > q) v = -1e30f;
        s[nf][j] = v;
        sm = fmaxf(sm, v);
      }
#pragma unroll
      for (int mk = 1; mk < 16; mk <<= 1)     // reduce within 16-lane half
        sm = fmaxf(sm, __shfl_xor(sm, mk, 32));
      float mnew = fmaxf(mrow[j], sm);
      float alpha = __expf(mrow[j] - mnew);
      mrow[j] = mnew;
      float rs = 0.f;
#pragma unroll
      for (int nf = 0; nf < 2; ++nf) {
        float pv = __expf(s[nf][j] - mnew);
        p[nf][j] = pv;
        rs += pv;
      }
#pragma unroll
      for (int mk = 1; mk < 16; mk <<= 1) rs += __shfl_xor(rs, mk, 32);
      lrow[j] = lrow[j] * alpha + rs;
#pragma unroll
      for (int df = 0; df < 4; ++df) o[df][j] *= alpha;
    }

    // Relayout P: C-layout -> A-layout via per-wave LDS scratch.
#pragma unroll
    for (int nf = 0; nf < 2; ++nf)
#pragma unroll
      for (int j = 0; j < 8; ++j)
        Ps[wave][j + hi * 8][nf * 16 + lane16] = (__bf16)p[nf][j];
    asm volatile("s_wait_dscnt 0x0" ::: "memory");

    v16bf aP;
    {
      int k0 = hi ? 8 : 0;
#pragma unroll
      for (int e = 0; e < 8; ++e) {
        aP[e]     = Ps[wave][lane16][k0 + e];
        aP[e + 8] = Ps[wave][lane16][16 + k0 + e];
      }
    }
    // O += P @ V : 4 d-frags, single K-step of 32.
#pragma unroll
    for (int df = 0; df < 4; ++df) {
      v16bf bv;
#pragma unroll
      for (int e = 0; e < 16; ++e) bv[e] = Vs[(e & 7) + hi * 8 + (e & 8) * 2][df * 16 + lane16];
      o[df] = wmma_bf16(aP, bv, o[df]);
    }
    __syncthreads();
  }

  // Normalize and store (B,T,H*64) bf16.
#pragma unroll
  for (int df = 0; df < 4; ++df)
#pragma unroll
    for (int j = 0; j < 8; ++j) {
      int t = qb + wave * 16 + j + hi * 8;
      int d = df * 16 + lane16;
      out[((size_t)(b * T + t)) * 256 + hd * 64 + d] = (__bf16)(o[df][j] / lrow[j]);
    }
}

// ---------------------------------------------------------------------------
// Host launcher
// ---------------------------------------------------------------------------
extern "C" void kernel_launch(void* const* d_in, const int* in_sizes, int n_in,
                              void* d_out, int out_size, void* d_ws, size_t ws_size,
                              hipStream_t stream) {
  (void)in_sizes; (void)n_in; (void)out_size; (void)ws_size;
  const float* x      = (const float*)d_in[0];   // (4,2048,4096)
  const float* W_down = (const float*)d_in[1];   // (256,4096)
  const float* W_up   = (const float*)d_in[2];   // (4096,256)
  const float* W_qkv  = (const float*)d_in[3];   // (768,256)
  const float* W_o    = (const float*)d_in[4];   // (256,256)
  const float* W_gate = (const float*)d_in[5];   // (512,256)
  const float* W_upff = (const float*)d_in[6];   // (512,256)
  const float* W_dff  = (const float*)d_in[7];   // (256,512)
  const float* g1     = (const float*)d_in[8];
  const float* g2     = (const float*)d_in[9];

  const int M = 8192;  // B*T

  char* ws = (char*)d_ws;
  size_t off = 0;
  auto alloc = [&](size_t bytes) -> void* {
    void* p = ws + off;
    off += (bytes + 255) & ~(size_t)255;
    return p;
  };
  __bf16* WtDown = (__bf16*)alloc((size_t)4096 * 256 * 2);
  __bf16* WtUp   = (__bf16*)alloc((size_t)256 * 4096 * 2);
  __bf16* WtQkv  = (__bf16*)alloc((size_t)256 * 768 * 2);
  __bf16* WtO    = (__bf16*)alloc((size_t)256 * 256 * 2);
  __bf16* WtGate = (__bf16*)alloc((size_t)256 * 512 * 2);
  __bf16* WtUpff = (__bf16*)alloc((size_t)256 * 512 * 2);
  __bf16* WtDff  = (__bf16*)alloc((size_t)512 * 256 * 2);
  float*  h      = (float*) alloc((size_t)M * 256 * 4);
  __bf16* hn     = (__bf16*)alloc((size_t)M * 256 * 2);
  __bf16* qkv    = (__bf16*)alloc((size_t)M * 768 * 2);
  __bf16* attn   = (__bf16*)alloc((size_t)M * 256 * 2);
  float*  h2     = (float*) alloc((size_t)M * 256 * 4);
  __bf16* hn2    = (__bf16*)alloc((size_t)M * 256 * 2);
  __bf16* ffin   = (__bf16*)alloc((size_t)M * 512 * 2);
  __bf16* h3     = (__bf16*)alloc((size_t)M * 256 * 2);

  auto wprep = [&](const float* W, __bf16* Wt, int N, int K) {
    wprep_kernel<<<dim3((N * K) / 256), dim3(256), 0, stream>>>(W, Wt, N, K);
  };
  wprep(W_down, WtDown, 256, 4096);
  wprep(W_up,   WtUp,   4096, 256);
  wprep(W_qkv,  WtQkv,  768, 256);
  wprep(W_o,    WtO,    256, 256);
  wprep(W_gate, WtGate, 512, 256);
  wprep(W_upff, WtUpff, 512, 256);
  wprep(W_dff,  WtDff,  256, 512);

  // h = x @ W_down^T   (M,4096)x(4096,256); wide 256-col tile: x read once.
  gemm_kernel<0, 0, 4><<<dim3(1, M / 128), 256, 0, stream>>>(
      x, WtDown, nullptr, h, M, 256, 4096);
  // hn = rmsnorm(h, g1) -> bf16
  rms_kernel<<<dim3(M / 8), 256, 0, stream>>>(h, g1, hn);
  // qkv = hn @ W_qkv^T -> bf16
  gemm_kernel<1, 1, 1><<<dim3(768 / 64, M / 128), 256, 0, stream>>>(
      hn, WtQkv, nullptr, qkv, M, 768, 256);
  // attention -> bf16 (B,T,256)
  attn_kernel<<<dim3(2048 / 128, 16), 256, 0, stream>>>(qkv, attn);
  // h2 = h + attn @ W_o^T -> fp32
  gemm_kernel<1, 2, 1><<<dim3(256 / 64, M / 128), 256, 0, stream>>>(
      attn, WtO, h, h2, M, 256, 256);
  // hn2 = rmsnorm(h2, g2) -> bf16
  rms_kernel<<<dim3(M / 8), 256, 0, stream>>>(h2, g2, hn2);
  // ffin = silu(hn2 @ W_gate^T) * (hn2 @ W_upff^T) -> bf16
  gemm_gateup_kernel<<<dim3(512 / 64, M / 128), 256, 0, stream>>>(
      hn2, WtGate, WtUpff, ffin, M, 512, 256);
  // h3 = bf16(h2 + ffin @ W_downff^T)
  gemm_kernel<1, 3, 1><<<dim3(256 / 64, M / 128), 256, 0, stream>>>(
      ffin, WtDff, h2, h3, M, 256, 512);
  // out = h3 @ W_up^T -> fp32 (B,T,4096)
  gemm_kernel<1, 0, 1><<<dim3(4096 / 64, M / 128), 256, 0, stream>>>(
      h3, WtUp, nullptr, (float*)d_out, M, 4096, 256);
}
